// MultiHeadAttention_29016799052649
// MI455X (gfx1250) — matrix operand order repaired and checked by reading, not compiled, not run on hardware
//
#include <hip/hip_runtime.h>
#include <hip/hip_bf16.h>
#include <stdint.h>

#define DMODEL 1024
#define SEQ    2048
#define NB     2
#define NH     16
#define DKH    64
#define MTOT   (NB*SEQ)   // 4096

typedef __attribute__((ext_vector_type(16))) __bf16 v16bf;
typedef __attribute__((ext_vector_type(8)))  __bf16 bf8v;
typedef __attribute__((ext_vector_type(2)))  __bf16 bf2;
typedef __attribute__((ext_vector_type(8)))  float  v8f;
typedef __attribute__((ext_vector_type(8)))  float  f8v;
typedef __attribute__((ext_vector_type(4)))  unsigned int u4;

union Frag { v16bf v; unsigned int u[8]; };

// pack two fp32 -> one dword of two bf16 (hardware cvt, RNE)
static __device__ __forceinline__ unsigned int pk2(float a, float b) {
  union { bf2 v; unsigned int u; } cv;
  cv.v.x = (__bf16)a; cv.v.y = (__bf16)b;
  return cv.u;
}
static __device__ __forceinline__ unsigned short bfr(float f) {
  union { __bf16 v; unsigned short u; } cv;
  cv.v = (__bf16)f;
  return cv.u;
}
static __device__ __forceinline__ v8f vzero() {
  v8f z = {0.f,0.f,0.f,0.f,0.f,0.f,0.f,0.f}; return z;
}
static __device__ __forceinline__ v8f wmma_bf16(v16bf a, v16bf b, v8f c) {
  return __builtin_amdgcn_wmma_f32_16x16x32_bf16(false, a, false, b, (short)0, c, false, false);
}

// A-matrix 16x32 bf16 per-lane fragment: chunks at elements half*8 and half*8+16
static __device__ __forceinline__ v16bf ldsplit_a(const unsigned short* p, int half) {
  Frag f;
  u4 c0 = *(const u4*)(p + half*8);
  u4 c1 = *(const u4*)(p + half*8 + 16);
  f.u[0]=c0.x; f.u[1]=c0.y; f.u[2]=c0.z; f.u[3]=c0.w;
  f.u[4]=c1.x; f.u[5]=c1.y; f.u[6]=c1.z; f.u[7]=c1.w;
  return f.v;
}
// B-matrix 32x16 bf16 per-lane fragment: chunks at elements half*8 and half*8+16
static __device__ __forceinline__ v16bf ldcontig_b(const unsigned short* p, int half) {
  Frag f;
  u4 c0 = *(const u4*)(p + half*8);
  u4 c1 = *(const u4*)(p + half*8 + 16);
  f.u[0]=c0.x; f.u[1]=c0.y; f.u[2]=c0.z; f.u[3]=c0.w;
  f.u[4]=c1.x; f.u[5]=c1.y; f.u[6]=c1.z; f.u[7]=c1.w;
  return f.v;
}

// ---------------- Kernel 0: bulk fp32 -> bf16 conversion --------------------
__global__ __launch_bounds__(256) void cvt_bf16_kernel(
    const float* __restrict__ src, __bf16* __restrict__ dst, int n)
{
  const int i = (blockIdx.x * 256 + threadIdx.x) * 8;
  if (i < n) {
    f8v v = *(const f8v*)(src + i);
    *(bf8v*)(dst + i) = __builtin_convertvector(v, bf8v);
  }
}

// ---------------- Kernel 1: QKV projections (y = x @ W^T), all-bf16 ---------
// Block = 8 waves; each wave computes a 32x64 tile (two 16-row A fragments
// sharing the four B fragments -> 1.5 loads per WMMA).
__global__ __launch_bounds__(256) void qkv_proj_kernel(
    const unsigned short* __restrict__ X, const unsigned short* __restrict__ W,
    unsigned short* __restrict__ Yrm, unsigned short* __restrict__ Yt,
    float scale)
{
  const int wave = threadIdx.x >> 5, lane = threadIdx.x & 31;
  const int half = lane >> 4, l16 = lane & 15;
  const int mbase = blockIdx.x * 256 + wave * 32;
  const int nbase = blockIdx.y * 64;

  v8f acc[2][4];
#pragma unroll
  for (int mt = 0; mt < 2; ++mt)
#pragma unroll
    for (int nt = 0; nt < 4; ++nt) acc[mt][nt] = vzero();

  const unsigned short* arow0 = X + (size_t)(mbase + l16) * DMODEL;
  const unsigned short* arow1 = arow0 + (size_t)16 * DMODEL;
  const unsigned short* brow  = W + (size_t)(nbase + l16) * DMODEL;

  for (int k = 0; k < DMODEL; k += 32) {
    __builtin_prefetch(arow0 + k + 128, 0, 1);
    __builtin_prefetch(arow1 + k + 128, 0, 1);
    v16bf a0 = ldsplit_a(arow0 + k, half);
    v16bf a1 = ldsplit_a(arow1 + k, half);
#pragma unroll
    for (int nt = 0; nt < 4; ++nt) {
      v16bf b = ldcontig_b(brow + (size_t)(nt * 16) * DMODEL + k, half);
      acc[0][nt] = wmma_bf16(a0, b, acc[0][nt]);
      acc[1][nt] = wmma_bf16(a1, b, acc[1][nt]);
    }
  }

  if (Yt) {  // transposed store: Yt[(b*NH+h)*DKH + n][s]
    const int b = mbase / SEQ;
#pragma unroll
    for (int mt = 0; mt < 2; ++mt) {
      const int srow = (mbase % SEQ) + mt * 16 + half * 8;
#pragma unroll
      for (int nt = 0; nt < 4; ++nt) {
        const int ncol = nbase + nt * 16 + l16;
        const int h = ncol >> 6, n = ncol & 63;
        u4 st;
        st.x = pk2(acc[mt][nt][0], acc[mt][nt][1]);
        st.y = pk2(acc[mt][nt][2], acc[mt][nt][3]);
        st.z = pk2(acc[mt][nt][4], acc[mt][nt][5]);
        st.w = pk2(acc[mt][nt][6], acc[mt][nt][7]);
        *(u4*)(Yt + ((size_t)((b * NH + h) * DKH + n)) * SEQ + srow) = st;
      }
    }
  } else {   // row-major store with optional scale (Q: log2e/sqrt(dk))
#pragma unroll
    for (int mt = 0; mt < 2; ++mt)
#pragma unroll
      for (int nt = 0; nt < 4; ++nt) {
        const int ncol = nbase + nt * 16 + l16;
#pragma unroll
        for (int j = 0; j < 8; ++j)
          Yrm[(size_t)(mbase + mt * 16 + j + half * 8) * DMODEL + ncol] =
              bfr(acc[mt][nt][j] * scale);
      }
  }
}

// ---------------- Kernel 2: causal flash attention (bf16 in/out) ------------
// One wave per 32-query tile (two 16-query subtiles of one head). S^T = K*Q^T
// puts softmax(P) directly in the A-layout for P*V; K and V fragments are
// loaded once and feed both subtiles -> 1.0 loads per WMMA.
static __device__ __forceinline__ Frag softmax_step(
    v8f& s0, v8f& s1, int q, int keyb, float& mrun, float& lrun,
    v8f o[4], int half)
{
  float mloc = -3.0e38f;
#pragma unroll
  for (int j = 0; j < 8; ++j) {
    if (keyb + j      > q) s0[j] = -3.0e38f;
    if (keyb + 16 + j > q) s1[j] = -3.0e38f;
    mloc = fmaxf(mloc, fmaxf(s0[j], s1[j]));
  }
  mloc = fmaxf(mloc, __shfl_xor(mloc, 16, 32));
  const float mnew = fmaxf(mrun, mloc);
  const float corr = __builtin_amdgcn_exp2f(mrun - mnew);

  Frag pf; float ls = 0.f;
#pragma unroll
  for (int j = 0; j < 4; ++j) {
    float p0 = __builtin_amdgcn_exp2f(s0[2*j]   - mnew);
    float p1 = __builtin_amdgcn_exp2f(s0[2*j+1] - mnew);
    float p2 = __builtin_amdgcn_exp2f(s1[2*j]   - mnew);
    float p3 = __builtin_amdgcn_exp2f(s1[2*j+1] - mnew);
    pf.u[j]     = pk2(p0, p1);
    pf.u[4 + j] = pk2(p2, p3);
    ls += (p0 + p1) + (p2 + p3);
  }
  ls += __shfl_xor(ls, 16, 32);
  lrun = lrun * corr + ls;
  mrun = mnew;
#pragma unroll
  for (int j = 0; j < 8; ++j) {
    const float cj = __shfl(corr, j + half * 8, 32);
    o[0][j] *= cj; o[1][j] *= cj; o[2][j] *= cj; o[3][j] *= cj;
  }
  return pf;
}

__global__ __launch_bounds__(256) void attn_kernel(
    const unsigned short* __restrict__ Qb,
    const unsigned short* __restrict__ Kb,
    const unsigned short* __restrict__ Vt,
    unsigned short* __restrict__ Xb)
{
  const int wave = threadIdx.x >> 5, lane = threadIdx.x & 31;
  const int half = lane >> 4, l16 = lane & 15;
  const int task = blockIdx.x * 8 + wave;       // b*NH*64 + h*64 + qt
  const int qt = task & 63;
  const int h  = (task >> 6) & 15;
  const int b  = task >> 10;
  const int q0 = qt * 32;

  const size_t qrow = ((size_t)(b * SEQ + q0 + l16)) * DMODEL + h * DKH;
  const v16bf qfA0 = ldcontig_b(Qb + qrow, half);                    // subtile A
  const v16bf qfA1 = ldcontig_b(Qb + qrow + 32, half);
  const v16bf qfB0 = ldcontig_b(Qb + qrow + (size_t)16 * DMODEL, half);      // subtile B
  const v16bf qfB1 = ldcontig_b(Qb + qrow + (size_t)16 * DMODEL + 32, half);

  v8f oA[4] = {vzero(), vzero(), vzero(), vzero()};
  v8f oB[4] = {vzero(), vzero(), vzero(), vzero()};
  float mrunA = -3.0e38f, lrunA = 0.f, mrunB = -3.0e38f, lrunB = 0.f;
  const int qA = q0 + l16, qB = q0 + 16 + l16;
  const int nkb = (q0 >> 5) + 1;                // causal: keys 0 .. q0+31
  const unsigned short* vhead = Vt + (size_t)((b * NH + h) * DKH) * SEQ;

  for (int kb = 0; kb < nkb; ++kb) {
    const int k0 = kb * 32;
    const unsigned short* kr0 = Kb + ((size_t)(b * SEQ + k0 + l16)) * DMODEL + h * DKH;
    const unsigned short* kr1 = kr0 + 16 * DMODEL;

    // K fragments loaded once, used by both subtiles
    const v16bf ka00 = ldsplit_a(kr0,      half);
    const v16bf ka01 = ldsplit_a(kr0 + 32, half);
    const v16bf ka10 = ldsplit_a(kr1,      half);
    const v16bf ka11 = ldsplit_a(kr1 + 32, half);

    v8f sA0 = vzero(), sA1 = vzero(), sB0 = vzero(), sB1 = vzero();
    sA0 = wmma_bf16(ka00, qfA0, sA0); sA0 = wmma_bf16(ka01, qfA1, sA0);
    sA1 = wmma_bf16(ka10, qfA0, sA1); sA1 = wmma_bf16(ka11, qfA1, sA1);
    sB0 = wmma_bf16(ka00, qfB0, sB0); sB0 = wmma_bf16(ka01, qfB1, sB0);
    sB1 = wmma_bf16(ka10, qfB0, sB1); sB1 = wmma_bf16(ka11, qfB1, sB1);

    const int keyb = k0 + half * 8;
    Frag pfA = softmax_step(sA0, sA1, qA, keyb, mrunA, lrunA, oA, half);
    Frag pfB = softmax_step(sB0, sB1, qB, keyb, mrunB, lrunB, oB, half);

    // V fragments loaded once, feed both subtiles
#pragma unroll
    for (int nt = 0; nt < 4; ++nt) {
      const unsigned short* vp = vhead + (size_t)(nt * 16 + l16) * SEQ + k0;
      const v16bf vf = ldcontig_b(vp, half);
      oA[nt] = wmma_bf16(pfA.v, vf, oA[nt]);
      oB[nt] = wmma_bf16(pfB.v, vf, oB[nt]);
    }
  }

  const float linvA = 1.0f / lrunA, linvB = 1.0f / lrunB;
#pragma unroll
  for (int j = 0; j < 8; ++j) {
    const float ljA = __shfl(linvA, j + half * 8, 32);
    const float ljB = __shfl(linvB, j + half * 8, 32);
    const size_t rowA = ((size_t)(b * SEQ + q0 + j + half * 8)) * DMODEL + h * DKH;
    const size_t rowB = rowA + (size_t)16 * DMODEL;
#pragma unroll
    for (int nt = 0; nt < 4; ++nt) {
      Xb[rowA + nt * 16 + l16] = bfr(oA[nt][j] * ljA);
      Xb[rowB + nt * 16 + l16] = bfr(oB[nt][j] * ljB);
    }
  }
}

// ---------------- Kernel 3: output projection (bf16 x @ Wo^T -> fp32) -------
__global__ __launch_bounds__(256) void out_proj_kernel(
    const unsigned short* __restrict__ Xb, const unsigned short* __restrict__ Wo,
    float* __restrict__ Out)
{
  const int wave = threadIdx.x >> 5, lane = threadIdx.x & 31;
  const int half = lane >> 4, l16 = lane & 15;
  const int mbase = blockIdx.x * 256 + wave * 32;
  const int nbase = blockIdx.y * 64;

  v8f acc[2][4];
#pragma unroll
  for (int mt = 0; mt < 2; ++mt)
#pragma unroll
    for (int nt = 0; nt < 4; ++nt) acc[mt][nt] = vzero();

  const unsigned short* arow0 = Xb + (size_t)(mbase + l16) * DMODEL;
  const unsigned short* arow1 = arow0 + (size_t)16 * DMODEL;
  const unsigned short* brow  = Wo + (size_t)(nbase + l16) * DMODEL;

  for (int k = 0; k < DMODEL; k += 32) {
    __builtin_prefetch(arow0 + k + 128, 0, 1);
    __builtin_prefetch(arow1 + k + 128, 0, 1);
    v16bf a0 = ldsplit_a(arow0 + k, half);
    v16bf a1 = ldsplit_a(arow1 + k, half);
#pragma unroll
    for (int nt = 0; nt < 4; ++nt) {
      v16bf bfrag = ldcontig_b(brow + (size_t)(nt * 16) * DMODEL + k, half);
      acc[0][nt] = wmma_bf16(a0, bfrag, acc[0][nt]);
      acc[1][nt] = wmma_bf16(a1, bfrag, acc[1][nt]);
    }
  }
#pragma unroll
  for (int mt = 0; mt < 2; ++mt)
#pragma unroll
    for (int nt = 0; nt < 4; ++nt) {
      const int ncol = nbase + nt * 16 + l16;
#pragma unroll
      for (int j = 0; j < 8; ++j)
        Out[(size_t)(mbase + mt * 16 + j + half * 8) * DMODEL + ncol] = acc[mt][nt][j];
    }
}

extern "C" void kernel_launch(void* const* d_in, const int* in_sizes, int n_in,
                              void* d_out, int out_size, void* d_ws, size_t ws_size,
                              hipStream_t stream) {
  const float* q  = (const float*)d_in[0];
  const float* k  = (const float*)d_in[1];
  const float* v  = (const float*)d_in[2];
  const float* Wq = (const float*)d_in[3];
  const float* Wk = (const float*)d_in[4];
  const float* Wv = (const float*)d_in[5];
  const float* Wo = (const float*)d_in[6];
  float* out = (float*)d_out;

  const size_t xel = (size_t)MTOT * DMODEL;     // 4M elems per activation
  const size_t wel = (size_t)DMODEL * DMODEL;   // 1M elems per weight
  unsigned short* ws = (unsigned short*)d_ws;
  unsigned short* Xq  = ws;                     // bf16 copies of inputs
  unsigned short* Xk  = Xq  + xel;
  unsigned short* Xv  = Xk  + xel;
  unsigned short* Wqb = Xv  + xel;              // bf16 weights
  unsigned short* Wkb = Wqb + wel;
  unsigned short* Wvb = Wkb + wel;
  unsigned short* Wob = Wvb + wel;
  unsigned short* Qb  = Wob + wel;              // [B*S, D], scaled by log2e/sqrt(dk)
  unsigned short* Kb  = Qb  + xel;              // [B*S, D]
  unsigned short* Vt  = Kb  + xel;              // [B*H*dk, S]
  unsigned short* Xb  = Vt  + xel;              // [B*S, D] attention output

  // pass 0: bulk fp32 -> bf16 (bandwidth bound, ~3.5us at 23.3 TB/s)
  const int gx = (int)(xel / (8 * 256)), gw = (int)(wel / (8 * 256));
  cvt_bf16_kernel<<<gx, 256, 0, stream>>>(q,  (__bf16*)Xq,  (int)xel);
  cvt_bf16_kernel<<<gx, 256, 0, stream>>>(k,  (__bf16*)Xk,  (int)xel);
  cvt_bf16_kernel<<<gx, 256, 0, stream>>>(v,  (__bf16*)Xv,  (int)xel);
  cvt_bf16_kernel<<<gw, 256, 0, stream>>>(Wq, (__bf16*)Wqb, (int)wel);
  cvt_bf16_kernel<<<gw, 256, 0, stream>>>(Wk, (__bf16*)Wkb, (int)wel);
  cvt_bf16_kernel<<<gw, 256, 0, stream>>>(Wv, (__bf16*)Wvb, (int)wel);
  cvt_bf16_kernel<<<gw, 256, 0, stream>>>(Wo, (__bf16*)Wob, (int)wel);

  dim3 gProj(MTOT / 256, DMODEL / 64);          // 32x64 tile per wave
  const float qscale = 0.18033688011112042f;    // (1/sqrt(64)) * log2(e)

  qkv_proj_kernel<<<gProj, 256, 0, stream>>>(Xq, Wqb, Qb, nullptr, qscale);
  qkv_proj_kernel<<<gProj, 256, 0, stream>>>(Xk, Wkb, Kb, nullptr, 1.0f);
  qkv_proj_kernel<<<gProj, 256, 0, stream>>>(Xv, Wvb, nullptr, Vt, 1.0f);
  attn_kernel<<<(NB * NH * (SEQ / 32)) / 8, 256, 0, stream>>>(Qb, Kb, Vt, Xb);
  out_proj_kernel<<<gProj, 256, 0, stream>>>(Xb, Wob, out);
}
